// LSTMxun_326417514598
// MI455X (gfx1250) — matrix operand order repaired and checked by reading, not compiled, not run on hardware
//
#include <hip/hip_runtime.h>

// ---------------------------------------------------------------------------
// Fused 4-layer LSTM (B=2048, T=512, IN=4, H=64, OUT=5) for gfx1250 (MI455X).
// One workgroup = 16 batch rows for all timesteps & layers. Recurrent and
// inter-layer GEMMs run on v_wmma_f32_16x16x32_f16 with weights pinned in
// VGPRs; activations/gates exchanged through LDS. x tile staged to LDS via
// the Tensor Data Mover (tensor_load_to_lds + s_wait_tensorcnt).
// ---------------------------------------------------------------------------

typedef __attribute__((ext_vector_type(16))) _Float16     v16h;
typedef __attribute__((ext_vector_type(8)))  float        v8f;
typedef __attribute__((ext_vector_type(4)))  int          v4i;
typedef __attribute__((ext_vector_type(8)))  int          v8i;
typedef __attribute__((ext_vector_type(4)))  unsigned int v4u;
typedef __attribute__((ext_vector_type(4)))  float        v4f;

#define B_    2048
#define T_    512
#define IN_   4
#define H_    64
#define L_    4
#define OUT_  5
#define G_    256      // 4*H gate width

// padded LDS strides (avoid bank conflicts)
#define WSTR  136      // halves per row of staged weight matrix (128 + 8)
#define HSTR  72       // halves per row of h state                (64 + 8)
#define GSTR  260      // floats per row of gate buffer            (256 + 4)

// dynamic LDS layout (bytes)
#define SM_WBUF  0
#define SM_X     (SM_WBUF + 256 * WSTR * 2)            // 69632
#define SM_G     (SM_X    + 16 * T_ * IN_ * 4)         // 200704
#define SM_H     (SM_G    + 16 * GSTR * 4)             // 217344
#define SM_B     (SM_H    + L_ * 16 * HSTR * 2)        // 226560
#define SM_TOT   (SM_B    + L_ * G_ * 4)               // 230656

#if __has_builtin(__builtin_amdgcn_tensor_load_to_lds) && \
    __has_builtin(__builtin_amdgcn_s_wait_tensorcnt)
#define HAVE_TDM 1
#endif

union FragU { v4i q[2]; v16h h; };

__device__ __forceinline__ v16h ld_frag(const _Float16* lo, const _Float16* hi) {
  FragU u;
  u.q[0] = *(const v4i*)lo;   // ds_load_b128
  u.q[1] = *(const v4i*)hi;   // ds_load_b128
  return u.h;
}

__device__ __forceinline__ v8f wmma(v16h a, v16h b, v8f c) {
  // D = A(16x32 f16) * B(32x16 f16) + C(16x16 f32)
  return __builtin_amdgcn_wmma_f32_16x16x32_f16(
      /*neg_a=*/false, a, /*neg_b=*/false, b,
      /*c_mod=*/(short)0, c, /*reuse_a=*/false, /*reuse_b=*/false);
}

__device__ __forceinline__ float sigmoidf_fast(float x) {
  return 1.0f / (1.0f + __expf(-x));
}
__device__ __forceinline__ float tanhf_fast(float x) {
  float e2 = __expf(2.0f * x);
  return (e2 - 1.0f) / (e2 + 1.0f);
}

__global__ __launch_bounds__(256, 1)
void lstm_fused_kernel(const float* __restrict__ x,
                       const float* __restrict__ Wih0,
                       const float* __restrict__ Wihr,
                       const float* __restrict__ Whh,
                       const float* __restrict__ bih,
                       const float* __restrict__ bhh,
                       const float* __restrict__ Wfc,
                       const float* __restrict__ bfc,
                       float* __restrict__ out) {
  extern __shared__ char smem[];
  _Float16* wbuf = (_Float16*)(smem + SM_WBUF);
  float*    xld  = (float*)   (smem + SM_X);
  float*    gbuf = (float*)   (smem + SM_G);
  _Float16* hbuf = (_Float16*)(smem + SM_H);
  float*    bsum = (float*)   (smem + SM_B);

  const int tid   = threadIdx.x;
  const int wv    = tid >> 5;          // wave 0..7
  const int lane  = tid & 31;
  const int lhalf = lane >> 4;         // 0: lanes 0-15, 1: lanes 16-31
  const int l16   = lane & 15;
  const int n0    = wv * 32;           // this wave's 32-col slice of gate dim
  const int row0  = blockIdx.x * 16;   // batch tile base

  // -------- stage x tile [16][T*IN] into LDS --------
#ifdef HAVE_TDM
  // Tensor Data Mover: 2D tile, 16 rows of 2048 f32, row stride T*IN floats.
  if (tid == 0) {
    const unsigned long long gaddr =
        (unsigned long long)(const void*)(x + (size_t)row0 * (T_ * IN_));
    v4u g0 = {};
    g0[0] = 1u;                                   // count=1, user descriptor
    g0[1] = (unsigned)SM_X;                       // lds_addr (bytes)
    g0[2] = (unsigned)(gaddr & 0xffffffffull);    // global_addr[31:0]
    g0[3] = (unsigned)((gaddr >> 32) & 0x01ffffffull) | (2u << 30); // type=2

    v8i g1 = {};
    g1[0] = (int)(2u << 16);                      // wg_mask=0, data_size=4B
    g1[1] = (int)(((unsigned)(T_ * IN_) & 0xffffu) << 16);  // tensor_dim0 lo16
    g1[2] = (int)(((unsigned)(T_ * IN_) >> 16) |
                  ((16u & 0xffffu) << 16));       // tensor_dim0 hi / dim1 lo
    g1[3] = (int)(((unsigned)(T_ * IN_) & 0xffffu) << 16);  // tile_dim0
    g1[4] = 16;                                   // tile_dim1 = 16 rows
    g1[5] = (int)(T_ * IN_);                      // tensor_dim0_stride lo32
    g1[6] = 0;
    g1[7] = 0;

    v4i z4 = {};
#if __clang_major__ >= 23
    v8i z8 = {};
    __builtin_amdgcn_tensor_load_to_lds(g0, g1, z4, z4, z8, 0);
#else
    __builtin_amdgcn_tensor_load_to_lds(g0, g1, z4, z4, 0);
#endif
    __builtin_amdgcn_s_wait_tensorcnt(0);
  }
#else
  for (int idx = tid; idx < 16 * T_; idx += 256) {
    const int r  = idx >> 9;           // T_*IN_/4 == 512 float4 per row
    const int t4 = idx & 511;
    ((v4f*)xld)[idx] =
        ((const v4f*)(x + (size_t)(row0 + r) * (T_ * IN_)))[t4];
  }
#endif
  // -------- fused bias --------
  for (int idx = tid; idx < L_ * G_; idx += 256)
    bsum[idx] = bih[idx] + bhh[idx];
  // -------- zero h state --------
  for (int idx = tid; idx < L_ * 16 * HSTR; idx += 256)
    hbuf[idx] = (_Float16)0.0f;
  __syncthreads();

  // -------- stage weights layer-by-layer, pin B-fragments in VGPRs --------
  // wfrag[chunkIdx][tile]; chunkIdx: layer0 -> 0..1, layer l>=1 -> 2+4*(l-1)..
  v16h wfrag[14][2];

  auto stage = [&](int l, int K) {
    const int tot = 256 * K;
    for (int idx = tid; idx < tot; idx += 256) {
      const int n = idx / K, k = idx - n * K;
      float w;
      if (l == 0)       w = Whh[n * H_ + k];
      else if (k < H_)  w = Wihr[((l - 1) * G_ + n) * H_ + k];
      else              w = Whh[(l * G_ + n) * H_ + (k - H_)];
      wbuf[n * WSTR + k] = (_Float16)w;
    }
  };
  auto grabB = [&](int tile, int chunk) -> v16h {
    // lane holds 16 contiguous K at chunk*32 + lhalf*16, column n0+tile*16+l16
    const _Float16* p = &wbuf[(n0 + tile * 16 + l16) * WSTR + chunk * 32 + lhalf * 16];
    return ld_frag(p, p + 8);
  };

  stage(0, 64);  __syncthreads();
  wfrag[0][0] = grabB(0, 0);  wfrag[0][1] = grabB(1, 0);
  wfrag[1][0] = grabB(0, 1);  wfrag[1][1] = grabB(1, 1);
  __syncthreads();
  stage(1, 128); __syncthreads();
  wfrag[2][0] = grabB(0, 0);  wfrag[2][1] = grabB(1, 0);
  wfrag[3][0] = grabB(0, 1);  wfrag[3][1] = grabB(1, 1);
  wfrag[4][0] = grabB(0, 2);  wfrag[4][1] = grabB(1, 2);
  wfrag[5][0] = grabB(0, 3);  wfrag[5][1] = grabB(1, 3);
  __syncthreads();
  stage(2, 128); __syncthreads();
  wfrag[6][0] = grabB(0, 0);  wfrag[6][1] = grabB(1, 0);
  wfrag[7][0] = grabB(0, 1);  wfrag[7][1] = grabB(1, 1);
  wfrag[8][0] = grabB(0, 2);  wfrag[8][1] = grabB(1, 2);
  wfrag[9][0] = grabB(0, 3);  wfrag[9][1] = grabB(1, 3);
  __syncthreads();
  stage(3, 128); __syncthreads();
  wfrag[10][0] = grabB(0, 0); wfrag[10][1] = grabB(1, 0);
  wfrag[11][0] = grabB(0, 1); wfrag[11][1] = grabB(1, 1);
  wfrag[12][0] = grabB(0, 2); wfrag[12][1] = grabB(1, 2);
  wfrag[13][0] = grabB(0, 3); wfrag[13][1] = grabB(1, 3);
  __syncthreads();

  // per-wave bias + layer-0 input weight registers
  float biasr[L_][2];
#pragma unroll
  for (int l = 0; l < L_; ++l) {
    biasr[l][0] = bsum[l * G_ + n0 + l16];
    biasr[l][1] = bsum[l * G_ + n0 + 16 + l16];
  }
  const v4f w0a = *(const v4f*)(Wih0 + (n0 + l16) * IN_);
  const v4f w0b = *(const v4f*)(Wih0 + (n0 + 16 + l16) * IN_);

  // A-matrix fragment: 16-bit 16x32, lane holds 16 contiguous K at col + lhalf*16
  auto grabA = [&](const _Float16* hlayer, int colchunk) -> v16h {
    const _Float16* p = hlayer + l16 * HSTR + colchunk * 32 + lhalf * 16;
    return ld_frag(p, p + 8);
  };

  // combine persona: thread owns column cj and 4 rows cr0..cr0+3
  const int cj  = tid & 63;
  const int cr0 = (tid >> 6) * 4;
  float creg[L_][4] = {};

  auto combine = [&](int l) {
    _Float16* hl = hbuf + l * 16 * HSTR;
#pragma unroll
    for (int rr = 0; rr < 4; ++rr) {
      const int r = cr0 + rr;
      const float gi = gbuf[r * GSTR + cj];
      const float gf = gbuf[r * GSTR + 64 + cj];
      const float gg = gbuf[r * GSTR + 128 + cj];
      const float go = gbuf[r * GSTR + 192 + cj];
      const float c  = sigmoidf_fast(gf) * creg[l][rr] +
                       sigmoidf_fast(gi) * tanhf_fast(gg);
      creg[l][rr] = c;
      hl[r * HSTR + cj] = (_Float16)(sigmoidf_fast(go) * tanhf_fast(c));
    }
  };

  auto store_g = [&](v8f a0, v8f a1) {
#pragma unroll
    for (int v = 0; v < 8; ++v) {
      const int m = v + lhalf * 8;
      gbuf[m * GSTR + n0 + l16]      = a0[v];
      gbuf[m * GSTR + n0 + 16 + l16] = a1[v];
    }
  };

  // ----------------------- time loop -----------------------
  for (int t = 0; t < T_; ++t) {
    // ---- layer 0: gates = x_t @ Wih0^T (VALU, K=4) + h0 @ Whh0^T (WMMA) ----
    {
      v8f acc0, acc1;
#pragma unroll
      for (int v = 0; v < 8; ++v) { acc0[v] = biasr[0][0]; acc1[v] = biasr[0][1]; }
#pragma unroll
      for (int v = 0; v < 8; ++v) {
        const int m = v + lhalf * 8;
        const v4f xr = *(const v4f*)&xld[(m * T_ + t) * IN_];
        acc0[v] += xr.x * w0a.x + xr.y * w0a.y + xr.z * w0a.z + xr.w * w0a.w;
        acc1[v] += xr.x * w0b.x + xr.y * w0b.y + xr.z * w0b.z + xr.w * w0b.w;
      }
      const _Float16* h0 = hbuf;                 // h_0(t-1)
      const v16h a0 = grabA(h0, 0), a1 = grabA(h0, 1);
      acc0 = wmma(a0, wfrag[0][0], acc0);
      acc0 = wmma(a1, wfrag[1][0], acc0);
      acc1 = wmma(a0, wfrag[0][1], acc1);
      acc1 = wmma(a1, wfrag[1][1], acc1);
      store_g(acc0, acc1);
    }
    __syncthreads();
    combine(0);
    __syncthreads();

#define LAYER_STEP(LIDX, BASE)                                            \
    {                                                                     \
      v8f acc0, acc1;                                                     \
      _Pragma("unroll")                                                   \
      for (int v = 0; v < 8; ++v) {                                       \
        acc0[v] = biasr[LIDX][0]; acc1[v] = biasr[LIDX][1];               \
      }                                                                   \
      const _Float16* hp = hbuf + (LIDX - 1) * 16 * HSTR; /* h_{l-1}(t) */\
      const _Float16* hc = hbuf + (LIDX)     * 16 * HSTR; /* h_l(t-1)  */ \
      const v16h a0 = grabA(hp, 0), a1 = grabA(hp, 1);                    \
      const v16h a2 = grabA(hc, 0), a3 = grabA(hc, 1);                    \
      acc0 = wmma(a0, wfrag[BASE + 0][0], acc0);                          \
      acc0 = wmma(a1, wfrag[BASE + 1][0], acc0);                          \
      acc0 = wmma(a2, wfrag[BASE + 2][0], acc0);                          \
      acc0 = wmma(a3, wfrag[BASE + 3][0], acc0);                          \
      acc1 = wmma(a0, wfrag[BASE + 0][1], acc1);                          \
      acc1 = wmma(a1, wfrag[BASE + 1][1], acc1);                          \
      acc1 = wmma(a2, wfrag[BASE + 2][1], acc1);                          \
      acc1 = wmma(a3, wfrag[BASE + 3][1], acc1);                          \
      store_g(acc0, acc1);                                                \
    }                                                                     \
    __syncthreads();                                                      \
    combine(LIDX);                                                        \
    __syncthreads();

    LAYER_STEP(1, 2)
    LAYER_STEP(2, 6)
    LAYER_STEP(3, 10)
#undef LAYER_STEP
  }

  // ----------------- final FC on h_3(T-1) -----------------
  __syncthreads();
  if (tid < 16 * OUT_) {
    const int r  = tid / OUT_;
    const int oc = tid - r * OUT_;
    const _Float16* h3 = hbuf + 3 * 16 * HSTR + r * HSTR;
    float acc = bfc[oc];
#pragma unroll 8
    for (int j = 0; j < H_; ++j)
      acc += (float)h3[j] * Wfc[oc * H_ + j];
    out[(size_t)(row0 + r) * OUT_ + oc] = acc;
  }
}

extern "C" void kernel_launch(void* const* d_in, const int* in_sizes, int n_in,
                              void* d_out, int out_size, void* d_ws, size_t ws_size,
                              hipStream_t stream) {
  (void)in_sizes; (void)n_in; (void)out_size; (void)d_ws; (void)ws_size;
  const float* x    = (const float*)d_in[0];
  const float* Wih0 = (const float*)d_in[1];
  const float* Wihr = (const float*)d_in[2];
  const float* Whh  = (const float*)d_in[3];
  const float* bih  = (const float*)d_in[4];
  const float* bhh  = (const float*)d_in[5];
  const float* Wfc  = (const float*)d_in[6];
  const float* bfc  = (const float*)d_in[7];
  float* out = (float*)d_out;

  static_assert(SM_TOT <= 320 * 1024, "exceeds WGP LDS");
  (void)hipFuncSetAttribute((const void*)lstm_fused_kernel,
                            hipFuncAttributeMaxDynamicSharedMemorySize,
                            (int)SM_TOT);

  lstm_fused_kernel<<<B_ / 16, 256, SM_TOT, stream>>>(
      x, Wih0, Wihr, Whh, bih, bhh, Wfc, bfc, out);
}
